// NGP_50405736185934
// MI455X (gfx1250) — matrix operand order repaired and checked by reading, not compiled, not run on hardware
//
#include <hip/hip_runtime.h>

typedef __attribute__((ext_vector_type(16))) _Float16 v16h;
typedef __attribute__((ext_vector_type(8)))  float    v8f;

#define NPTS   524288
#define NLVL   8
#define TSZ    524288u
#define TMASK  (TSZ - 1u)
#define FEAT   16
#define U1     64
#define U2     32
#define NHEAD  2
#define BLK    256

union FragA { v16h v; unsigned u[8]; };

__device__ __forceinline__ unsigned pack2(_Float16 a, _Float16 b) {
    union { _Float16 h[2]; unsigned u; } t;
    t.h[0] = a; t.h[1] = b;
    return t.u;
}

// Build an A-matrix fragment (16x32 f16, MxK) from a row-major LDS matrix.
// Layout: lanes0-15 M=lane, VGPR v: K={2v,2v+1} (K 0..15)
//         lanes16-31: same M, K offset +16 (K 16..31)
__device__ __forceinline__ v16h buildA(const _Float16* src, int stride, int m,
                                       int hi8, int kreal) {
    FragA a;
    #pragma unroll
    for (int v = 0; v < 8; ++v) {
        int k = (hi8 << 1) + (v << 1);
        int kc = (k < kreal) ? k : 0;                 // keep address in-bounds
        unsigned val = *(const unsigned*)&src[m * stride + kc];
        a.u[v] = (k < kreal) ? val : 0u;              // zero-pad K
    }
    return a.v;
}

// Load a pre-swizzled B fragment: 8 contiguous dwords per lane -> 2x ds_load_b128
__device__ __forceinline__ v16h loadB(const unsigned* fragbuf, int lane) {
    FragA b;
    const unsigned* p = fragbuf + lane * 8;
    #pragma unroll
    for (int v = 0; v < 8; ++v) b.u[v] = p[v];
    return b.v;
}

__device__ __forceinline__ float leaky(float x) {
    return fmaxf(x, 0.01f * x);   // branchless leaky_relu(0.01)
}

__global__ __launch_bounds__(BLK) void ngp_fused(
    const float* __restrict__ x,
    const float* __restrict__ levels,
    const float* __restrict__ grid,
    const float* __restrict__ w1, const float* __restrict__ b1,
    const float* __restrict__ w2, const float* __restrict__ b2,
    const float* __restrict__ w3, const float* __restrict__ b3,
    float* __restrict__ out)
{
    __shared__ _Float16 s_feat[BLK * FEAT];            //  8 KB encoded features
    __shared__ _Float16 s_h1[BLK * U1];                // 32 KB layer-1 acts
    __shared__ _Float16 s_h2[BLK * U2];                // 16 KB layer-2 acts
    // Weight fragments pre-swizzled to WMMA B layout: [frag][lane][8 dwords]
    __shared__ __align__(16) unsigned s_fw1[4 * 256];  //  4 KB (w1, K pad 16->32)
    __shared__ __align__(16) unsigned s_fw2[4 * 256];  //  4 KB (w2, 2 nt x 2 kc)
    __shared__ __align__(16) unsigned s_fw3[256];      //  1 KB (w3, N pad 2->16)
    __shared__ float s_b1[U1];
    __shared__ float s_b2[U2];
    __shared__ float s_b3v[NHEAD];

    const int tid = threadIdx.x;

    // ---- stage 0: weights -> LDS, swizzled directly into B-fragment layout ----
    // fragment dword u: lane = (u>>3)&31, v = u&7; per-lane dword packs f16 pair
    // (K = kbase+2v, kbase+2v+1) at column N; kbase = 16 for lanes 16-31.
    for (int u = tid; u < 2304; u += BLK) {
        const int lane_ = (u >> 3) & 31;
        const int v     = u & 7;
        const int nlo_  = lane_ & 15;
        const bool hi_  = lane_ >= 16;
        if (u < 1024) {                       // w1: 4 N-tiles, K real = 16
            const int nt  = u >> 8;
            const int col = nt * 16 + nlo_;
            const int k   = 2 * v;            // lo-lane K; hi lanes = zero pad
            const unsigned val =
                pack2((_Float16)w1[k * U1 + col], (_Float16)w1[(k + 1) * U1 + col]);
            s_fw1[u] = hi_ ? 0u : val;
        } else if (u < 2048) {                // w2: 2 N-tiles x 2 K-chunks
            const int i2  = u - 1024;
            const int nt  = i2 >> 9;
            const int c   = (i2 >> 8) & 1;
            const int col = nt * 16 + nlo_;
            const int k   = c * 32 + (hi_ ? 16 : 0) + 2 * v;
            s_fw2[i2] =
                pack2((_Float16)w2[k * U2 + col], (_Float16)w2[(k + 1) * U2 + col]);
        } else {                              // w3: K = 32 exact, N padded 2->16
            const int k = (hi_ ? 16 : 0) + 2 * v;
            const _Float16 a = (nlo_ < NHEAD) ? (_Float16)w3[k * NHEAD + nlo_]
                                              : (_Float16)0.0f;
            const _Float16 b = (nlo_ < NHEAD) ? (_Float16)w3[(k + 1) * NHEAD + nlo_]
                                              : (_Float16)0.0f;
            s_fw3[u - 2048] = pack2(a, b);
        }
    }
    if (tid < U1)    s_b1[tid]  = b1[tid];
    if (tid < U2)    s_b2[tid]  = b2[tid];
    if (tid < NHEAD) s_b3v[tid] = b3[tid];

    // ---- stage 1: multiresolution hash-grid encode, one point per thread ----
    const int p = blockIdx.x * BLK + tid;
    const float px = x[3 * p + 0], py = x[3 * p + 1], pz = x[3 * p + 2];
    const float2* __restrict__ g2 = (const float2*)grid;
    #pragma unroll
    for (int l = 0; l < NLVL; ++l) {
        const float s = levels[l];
        const float xs0 = px * s, xs1 = py * s, xs2 = pz * s;
        const float f0 = floorf(xs0), f1 = floorf(xs1), f2 = floorf(xs2);
        const float wx = xs0 - f0, wy = xs1 - f1, wz = xs2 - f2;
        const unsigned cx = (unsigned)f0, cy = (unsigned)f1, cz = (unsigned)f2;
        float a0 = 0.0f, a1 = 0.0f;
        #pragma unroll
        for (int c = 0; c < 8; ++c) {
            const unsigned i = (c >> 2) & 1u, j = (c >> 1) & 1u, k = c & 1u;
            const unsigned h = ((cx + i) + (cy + j) * 2654435761u
                                         + (cz + k) * 805459861u) & TMASK;
            const float2 v = g2[(size_t)l * TSZ + h];     // L2-resident gather
            const float cw = (i ? wx : 1.0f - wx)
                           * (j ? wy : 1.0f - wy)
                           * (k ? wz : 1.0f - wz);
            a0 += v.x * cw;
            a1 += v.y * cw;
        }
        s_feat[tid * FEAT + 2 * l + 0] = (_Float16)a0;
        s_feat[tid * FEAT + 2 * l + 1] = (_Float16)a1;
    }
    __syncthreads();

    // ---- stage 2: MLP via v_wmma_f32_16x16x32_f16, 2 M-tiles per wave ----
    const int lane = tid & 31;
    const int wave = tid >> 5;
    const int nlo  = lane & 15;
    const bool hi  = lane >= 16;
    const int hi8  = hi ? 8 : 0;
    const int m    = lane & 15;

    // B fragments: contiguous vector loads from the pre-swizzled buffers
    v16h bL1[4], bL2[2][2], bL3;
    #pragma unroll
    for (int nt = 0; nt < 4; ++nt) bL1[nt] = loadB(s_fw1 + nt * 256, lane);
    #pragma unroll
    for (int nt = 0; nt < 2; ++nt)
        #pragma unroll
        for (int c = 0; c < 2; ++c)
            bL2[nt][c] = loadB(s_fw2 + (nt * 2 + c) * 256, lane);
    bL3 = loadB(s_fw3, lane);

    const float b3val = s_b3v[nlo & 1];   // valid for nlo<2; other lanes unused

    #pragma unroll
    for (int t = 0; t < 2; ++t) {
        const int base = (wave * 2 + t) * 16;

        // layer 1: [16 x 16] @ [16 x 64]  (K padded to 32), bias seeded in C
        const v16h aF = buildA(s_feat + base * FEAT, FEAT, m, hi8, FEAT);
        #pragma unroll
        for (int nt = 0; nt < 4; ++nt) {
            const int no = nt * 16 + nlo;
            const float bias = s_b1[no];
            v8f c;
            #pragma unroll
            for (int r = 0; r < 8; ++r) c[r] = bias;
            c = __builtin_amdgcn_wmma_f32_16x16x32_f16(false, aF, false, bL1[nt],
                                                       (short)0, c, false, false);
            #pragma unroll
            for (int r = 0; r < 8; ++r) {
                const int mo = r + (hi ? 8 : 0);
                s_h1[(base + mo) * U1 + no] = (_Float16)leaky(c[r]);
            }
        }

        // layer 2: [16 x 64] @ [64 x 32]  (2 K-chunks accumulate)
        const v16h aH0 = buildA(s_h1 + base * U1,      U1, m, hi8, 32);
        const v16h aH1 = buildA(s_h1 + base * U1 + 32, U1, m, hi8, 32);
        #pragma unroll
        for (int nt = 0; nt < 2; ++nt) {
            const int no = nt * 16 + nlo;
            const float bias = s_b2[no];
            v8f c;
            #pragma unroll
            for (int r = 0; r < 8; ++r) c[r] = bias;
            c = __builtin_amdgcn_wmma_f32_16x16x32_f16(false, aH0, false, bL2[nt][0],
                                                       (short)0, c, false, false);
            c = __builtin_amdgcn_wmma_f32_16x16x32_f16(false, aH1, false, bL2[nt][1],
                                                       (short)0, c, false, false);
            #pragma unroll
            for (int r = 0; r < 8; ++r) {
                const int mo = r + (hi ? 8 : 0);
                s_h2[(base + mo) * U2 + no] = (_Float16)leaky(c[r]);
            }
        }

        // layer 3: [16 x 32] @ [32 x 2]  (N padded to 16), bias seeded in C
        const v16h aH2 = buildA(s_h2 + base * U2, U2, m, hi8, 32);
        v8f c3;
        #pragma unroll
        for (int r = 0; r < 8; ++r) c3[r] = b3val;
        c3 = __builtin_amdgcn_wmma_f32_16x16x32_f16(false, aH2, false, bL3,
                                                    (short)0, c3, false, false);
        if (nlo < NHEAD) {
            const int gbase = blockIdx.x * BLK + base;
            #pragma unroll
            for (int r = 0; r < 8; ++r) {
                const int mo = r + (hi ? 8 : 0);
                out[(gbase + mo) * NHEAD + nlo] = c3[r];
            }
        }
    }
}

extern "C" void kernel_launch(void* const* d_in, const int* in_sizes, int n_in,
                              void* d_out, int out_size, void* d_ws, size_t ws_size,
                              hipStream_t stream) {
    const float* x      = (const float*)d_in[0];
    const float* levels = (const float*)d_in[1];
    const float* grid   = (const float*)d_in[2];
    const float* w1     = (const float*)d_in[3];
    const float* b1     = (const float*)d_in[4];
    const float* w2     = (const float*)d_in[5];
    const float* b2     = (const float*)d_in[6];
    const float* w3     = (const float*)d_in[7];
    const float* b3     = (const float*)d_in[8];
    float* out          = (float*)d_out;

    hipLaunchKernelGGL(ngp_fused, dim3(NPTS / BLK), dim3(BLK), 0, stream,
                       x, levels, grid, w1, b1, w2, b2, w3, b3, out);
}
